// MultiHeadAttention_16037407883269
// MI455X (gfx1250) — matrix operand order repaired and checked by reading, not compiled, not run on hardware
//
#include <hip/hip_runtime.h>
#include <hip/hip_bf16.h>

typedef __attribute__((ext_vector_type(16))) _Float16 v16h;
typedef __attribute__((ext_vector_type(8)))  float    v8f;

#define S_LEN  2048
#define DMODEL 1024
#define NHEAD  16
#define DHEAD  64

// ---------------------------------------------------------------------------
// WMMA helpers (CDNA5 gfx1250, wave32). D = A(16x32 f16) * B(32x16 f16) + C(f32)
// A layout: lane l -> row l%16; k = 8*(l/16) + {0..7} for elems 0..7,
//                              k = 8*(l/16) + 16 + {0..7} for elems 8..15.
// B layout: lane l -> col l%16; same k order as A.
// C layout: vgpr g, lane l -> row g + 8*(l/16), col l%16.
// ---------------------------------------------------------------------------
__device__ __forceinline__ v8f wmma_f16(v16h a, v16h b, v8f c) {
  return __builtin_amdgcn_wmma_f32_16x16x32_f16(false, a, false, b, (short)0, c,
                                                false, false);
}

__device__ __forceinline__ v16h a_frag_f32(const float* row, int kk, int half) {
  const float* p0 = row + kk + half * 8;
  v16h a;
#pragma unroll
  for (int i = 0; i < 8; ++i) {
    a[i]     = (_Float16)p0[i];
    a[8 + i] = (_Float16)p0[16 + i];
  }
  return a;
}

__device__ __forceinline__ v16h a_frag_f16(const _Float16* row, int kk, int half) {
  const _Float16* p0 = row + kk + half * 8;
  v16h a;
#pragma unroll
  for (int i = 0; i < 8; ++i) {
    a[i]     = p0[i];
    a[8 + i] = p0[16 + i];
  }
  return a;
}

// B[k][n] fragment where src is stored [n][k] row-major with leading dim ldk.
__device__ __forceinline__ v16h b_frag(const _Float16* src, int ldk, int n0,
                                       int koff, int lane) {
  const _Float16* p = src + (size_t)(n0 + (lane & 15)) * ldk + koff +
                      ((lane >> 4) & 1) * 8;
  typedef __attribute__((ext_vector_type(8))) _Float16 v8h_;
  v8h_ lo = *(const v8h_*)p;
  v8h_ hi = *(const v8h_*)(p + 16);
  return __builtin_shufflevector(lo, hi, 0, 1, 2, 3, 4, 5, 6, 7, 8, 9, 10, 11, 12, 13, 14, 15);
}

// ---------------------------------------------------------------------------
// Kernel 0: fp32 weights [K][N] -> f16 transposed [N][K]
// ---------------------------------------------------------------------------
__global__ __launch_bounds__(256) void pack_weights(
    const float* __restrict__ Wq, const float* __restrict__ Wk,
    const float* __restrict__ Wv, const float* __restrict__ Wo,
    _Float16* __restrict__ WqT, _Float16* __restrict__ WkT,
    _Float16* __restrict__ WvT, _Float16* __restrict__ WoT) {
  int idx = blockIdx.x * 256 + threadIdx.x;  // [0, 4*1024*1024)
  int w   = idx >> 20;
  int rem = idx & 0xFFFFF;
  int n   = rem >> 10;
  int k   = rem & 1023;
  const float* src = (w == 0) ? Wq : (w == 1) ? Wk : (w == 2) ? Wv : Wo;
  _Float16*    dst = (w == 0) ? WqT : (w == 1) ? WkT : (w == 2) ? WvT : WoT;
  dst[(size_t)n * DMODEL + k] = (_Float16)src[(size_t)k * DMODEL + n];
}

// ---------------------------------------------------------------------------
// Kernel 1: fused QKV projection. One wave per 16x64 tile (one full head).
// gridDim = (256 row-tiles, 48 = {Q,K,V} x 16 head-tiles)
// Q,K -> [B,H,S,DH] f16 ; V -> transposed [B,H,DH,S] f16
// ---------------------------------------------------------------------------
__global__ __launch_bounds__(32) void qkv_proj(
    const float* __restrict__ Q, const float* __restrict__ K,
    const float* __restrict__ V, const _Float16* __restrict__ WqT,
    const _Float16* __restrict__ WkT, const _Float16* __restrict__ WvT,
    const float* __restrict__ bq, const float* __restrict__ bk,
    const float* __restrict__ bv, _Float16* __restrict__ Qh,
    _Float16* __restrict__ Kh, _Float16* __restrict__ Vt) {
  int lane = threadIdx.x;
  int half = (lane >> 4) & 1, l16 = lane & 15;
  int m0    = blockIdx.x * 16;
  int which = blockIdx.y >> 4;         // 0=Q 1=K 2=V
  int h     = blockIdx.y & 15;         // head -> columns h*64 .. h*64+63
  int n0    = h * DHEAD;

  const float*    X    = (which == 0) ? Q : (which == 1) ? K : V;
  const _Float16* Wt   = (which == 0) ? WqT : (which == 1) ? WkT : WvT;
  const float*    bias = (which == 0) ? bq : (which == 1) ? bk : bv;
  const float*    arow = X + (size_t)(m0 + l16) * DMODEL;

  v8f c[4];
#pragma unroll
  for (int t = 0; t < 4; ++t) c[t] = (v8f){};
  for (int kk = 0; kk < DMODEL; kk += 32) {
    v16h a = a_frag_f32(arow, kk, half);  // reused across 4 column sub-tiles
#pragma unroll
    for (int t = 0; t < 4; ++t) {
      v16h b = b_frag(Wt, DMODEL, n0 + t * 16, kk, lane);
      c[t] = wmma_f16(a, b, c[t]);
    }
  }

#pragma unroll
  for (int t = 0; t < 4; ++t) {
    float bval = bias[n0 + t * 16 + l16];
    int   dh   = t * 16 + l16;
#pragma unroll
    for (int g = 0; g < 8; ++g) {
      int      m  = m0 + g + half * 8;
      int      b_ = m >> 11;  // / S_LEN
      int      s  = m & 2047;
      _Float16 hv = (_Float16)(c[t][g] + bval);
      size_t   bh = (size_t)(b_ * NHEAD + h);
      if (which == 2) {
        Vt[(bh * DHEAD + dh) * S_LEN + s] = hv;
      } else {
        _Float16* dst = (which == 0) ? Qh : Kh;
        dst[(bh * S_LEN + s) * DHEAD + dh] = hv;
      }
    }
  }
}

// ---------------------------------------------------------------------------
// Kernel 2: flash attention, one wave per (b,h, 32-row q-tile): two 16-row
// tiles share every K/V fragment. gridDim = (S/32 = 64, B*H = 32).
// Causal mask computed analytically (mask input is triu(ones,1)).
// ---------------------------------------------------------------------------
__global__ __launch_bounds__(32) void flash_attn(
    const _Float16* __restrict__ Qh, const _Float16* __restrict__ Kh,
    const _Float16* __restrict__ Vt, _Float16* __restrict__ AO) {
  __shared__ _Float16 P[32 * 32];  // probs tile, row-major [32][32]
  int lane = threadIdx.x;
  int half = (lane >> 4) & 1, l16 = lane & 15;
  int q0 = blockIdx.x * 32;
  int bh = blockIdx.y;

  const _Float16* Qb = Qh + (size_t)bh * S_LEN * DHEAD;
  const _Float16* Kb = Kh + (size_t)bh * S_LEN * DHEAD;
  const _Float16* Vb = Vt + (size_t)bh * DHEAD * S_LEN;

  v16h aq[2][2];
#pragma unroll
  for (int rt = 0; rt < 2; ++rt) {
    const _Float16* qrow = Qb + (size_t)(q0 + rt * 16 + l16) * DHEAD;
    aq[rt][0] = a_frag_f16(qrow, 0, half);
    aq[rt][1] = a_frag_f16(qrow, 32, half);
  }

  v8f   acc[2][4];
  float mrow[2][8], lrow[2][8];
#pragma unroll
  for (int rt = 0; rt < 2; ++rt) {
#pragma unroll
    for (int t = 0; t < 4; ++t) acc[rt][t] = (v8f){};
#pragma unroll
    for (int g = 0; g < 8; ++g) { mrow[rt][g] = -3.0e38f; lrow[rt][g] = 0.f; }
  }
  const float scale = 0.125f;  // 1/sqrt(DH)

  for (int j = 0; j <= q0 + 31; j += 32) {  // skip fully-masked key chunks
    // K fragments for cols j..j+15 (ct=0) and j+16..j+31 (ct=1), K-dim = 64
    v16h bk[2][2];
#pragma unroll
    for (int ct = 0; ct < 2; ++ct) {
      bk[ct][0] = b_frag(Kb, DHEAD, j + ct * 16, 0, lane);
      bk[ct][1] = b_frag(Kb, DHEAD, j + ct * 16, 32, lane);
    }
#pragma unroll
    for (int rt = 0; rt < 2; ++rt) {
      v8f s0 = (v8f){}, s1 = (v8f){};
      s0 = wmma_f16(aq[rt][0], bk[0][0], s0);
      s0 = wmma_f16(aq[rt][1], bk[0][1], s0);
      s1 = wmma_f16(aq[rt][0], bk[1][0], s1);
      s1 = wmma_f16(aq[rt][1], bk[1][1], s1);
#pragma unroll
      for (int g = 0; g < 8; ++g) {
        int   row = q0 + rt * 16 + g + half * 8;
        float x1  = s0[g] * scale; if (j + l16 > row)      x1 -= 1e9f;
        float x2  = s1[g] * scale; if (j + 16 + l16 > row) x2 -= 1e9f;
        // row max across the 16 lanes holding this row
        float rm = fmaxf(x1, x2);
        rm = fmaxf(rm, __shfl_xor(rm, 1, 16));
        rm = fmaxf(rm, __shfl_xor(rm, 2, 16));
        rm = fmaxf(rm, __shfl_xor(rm, 4, 16));
        rm = fmaxf(rm, __shfl_xor(rm, 8, 16));
        float mnew = fmaxf(mrow[rt][g], rm);
        float corr = __expf(mrow[rt][g] - mnew);
        float p1   = __expf(x1 - mnew);
        float p2   = __expf(x2 - mnew);
        float rs   = p1 + p2;
        rs += __shfl_xor(rs, 1, 16);
        rs += __shfl_xor(rs, 2, 16);
        rs += __shfl_xor(rs, 4, 16);
        rs += __shfl_xor(rs, 8, 16);
        lrow[rt][g] = lrow[rt][g] * corr + rs;
        mrow[rt][g] = mnew;
        acc[rt][0][g] *= corr; acc[rt][1][g] *= corr;
        acc[rt][2][g] *= corr; acc[rt][3][g] *= corr;
        int mloc = rt * 16 + g + half * 8;
        P[mloc * 32 + l16]      = (_Float16)p1;  // C-layout -> LDS (transpose)
        P[mloc * 32 + 16 + l16] = (_Float16)p2;
      }
    }
    __syncthreads();  // single-wave WG: cheap; forces s_wait_dscnt ordering

    v16h bv[4];
#pragma unroll
    for (int t = 0; t < 4; ++t)
      bv[t] = b_frag(Vb, S_LEN, t * 16, j, lane);  // Vt rows = dh, contig seq
#pragma unroll
    for (int rt = 0; rt < 2; ++rt) {
      v16h ap = a_frag_f16(&P[(size_t)(rt * 16 + l16) * 32], 0, half);
#pragma unroll
      for (int t = 0; t < 4; ++t) acc[rt][t] = wmma_f16(ap, bv[t], acc[rt][t]);
    }
    __syncthreads();
  }

  int b_ = bh >> 4, h = bh & 15;
#pragma unroll
  for (int rt = 0; rt < 2; ++rt) {
#pragma unroll
    for (int g = 0; g < 8; ++g) {
      int    s   = q0 + rt * 16 + g + half * 8;
      float  inv = 1.0f / lrow[rt][g];
      size_t rb  = ((size_t)(b_ * S_LEN + s)) * (NHEAD * DHEAD) + h * DHEAD;
      AO[rb + 0  + l16] = (_Float16)(acc[rt][0][g] * inv);
      AO[rb + 16 + l16] = (_Float16)(acc[rt][1][g] * inv);
      AO[rb + 32 + l16] = (_Float16)(acc[rt][2][g] * inv);
      AO[rb + 48 + l16] = (_Float16)(acc[rt][3][g] * inv);
    }
  }
}

// ---------------------------------------------------------------------------
// Kernel 3: output projection [4096,1024]f16 x WoT + bo -> fp32 d_out
// One wave per 16x64 tile. gridDim = (256, 16).
// ---------------------------------------------------------------------------
__global__ __launch_bounds__(32) void out_proj(
    const _Float16* __restrict__ AO, const _Float16* __restrict__ WoT,
    const float* __restrict__ bo, float* __restrict__ out) {
  int lane = threadIdx.x;
  int half = (lane >> 4) & 1, l16 = lane & 15;
  int m0 = blockIdx.x * 16, n0 = blockIdx.y * 64;
  const _Float16* arow = AO + (size_t)(m0 + l16) * DMODEL;
  v8f c[4];
#pragma unroll
  for (int t = 0; t < 4; ++t) c[t] = (v8f){};
  for (int kk = 0; kk < DMODEL; kk += 32) {
    v16h a = a_frag_f16(arow, kk, half);  // reused across 4 column sub-tiles
#pragma unroll
    for (int t = 0; t < 4; ++t) {
      v16h b = b_frag(WoT, DMODEL, n0 + t * 16, kk, lane);
      c[t] = wmma_f16(a, b, c[t]);
    }
  }
#pragma unroll
  for (int t = 0; t < 4; ++t) {
    float bias = bo[n0 + t * 16 + l16];
#pragma unroll
    for (int g = 0; g < 8; ++g)
      out[(size_t)(m0 + g + half * 8) * DMODEL + n0 + t * 16 + l16] =
          c[t][g] + bias;
  }
}

// ---------------------------------------------------------------------------
extern "C" void kernel_launch(void* const* d_in, const int* in_sizes, int n_in,
                              void* d_out, int out_size, void* d_ws,
                              size_t ws_size, hipStream_t stream) {
  (void)in_sizes; (void)n_in; (void)out_size; (void)ws_size;
  const float* Q  = (const float*)d_in[0];
  const float* K  = (const float*)d_in[1];
  const float* V  = (const float*)d_in[2];
  // d_in[3] = attn_mask (causal triu) — applied analytically in flash_attn
  const float* Wq = (const float*)d_in[4];
  const float* bq = (const float*)d_in[5];
  const float* Wk = (const float*)d_in[6];
  const float* bk = (const float*)d_in[7];
  const float* Wv = (const float*)d_in[8];
  const float* bv = (const float*)d_in[9];
  const float* Wo = (const float*)d_in[10];
  const float* bo = (const float*)d_in[11];
  float*       out = (float*)d_out;

  char*        ws = (char*)d_ws;
  const size_t MB = 1024 * 1024;
  _Float16* WqT = (_Float16*)(ws + 0 * MB);
  _Float16* WkT = (_Float16*)(ws + 2 * MB);
  _Float16* WvT = (_Float16*)(ws + 4 * MB);
  _Float16* WoT = (_Float16*)(ws + 6 * MB);
  _Float16* Qh  = (_Float16*)(ws + 8 * MB);   // [B,H,S,DH]
  _Float16* Kh  = (_Float16*)(ws + 16 * MB);  // [B,H,S,DH]
  _Float16* Vt  = (_Float16*)(ws + 24 * MB);  // [B,H,DH,S]
  _Float16* AO  = (_Float16*)(ws + 32 * MB);  // [B*S, H*DH]

  pack_weights<<<dim3(4 * 1024 * 1024 / 256), dim3(256), 0, stream>>>(
      Wq, Wk, Wv, Wo, WqT, WkT, WvT, WoT);
  qkv_proj<<<dim3(256, 48), dim3(32), 0, stream>>>(
      Q, K, V, WqT, WkT, WvT, bq, bk, bv, Qh, Kh, Vt);
  flash_attn<<<dim3(64, 32), dim3(32), 0, stream>>>(Qh, Kh, Vt, AO);
  out_proj<<<dim3(256, 16), dim3(32), 0, stream>>>(AO, WoT, bo, out);
}
